// Attention_28948079575569
// MI455X (gfx1250) — compile-verified, matrix operand order checked by reading, not run on hardware
//
#include <hip/hip_runtime.h>

typedef __attribute__((ext_vector_type(16))) _Float16 v16h;
typedef __attribute__((ext_vector_type(8)))  float    v8f;
typedef __attribute__((ext_vector_type(4)))  _Float16 v4h;

#define DIM    512
#define RPB    128                 // rows (time steps) per workgroup
#define MT     8                   // RPB/16 m-tiles
#define KT     16                  // DIM/32 k-steps
#define BATCH  32
#define TLEN   4096
#define NCHUNK (TLEN / RPB)        // 32 chunks per batch
#define NWG    (BATCH * NCHUNK)    // 1024 workgroups

// fragment arena index: kt-major so all 8 m-tiles of one kt step fit in 8KB
// (pure immediate ds offsets inside the K loop)
#define FRAG_IDX(kt, mt, ln) ((((kt) * MT + (mt)) * 32 + (ln)) * 16)

// ---------------------------------------------------------------------------
// Kernel 0: pack W (f32 [E][K] row-major) into f16 WMMA B-fragment layout.
//   fragment (nt,kt): 32 lanes x 16 halves; n = nt*16 + lane%16,
//   k = kt*32 + (j&7) + (j>=8 ? 16:0) + (lane>=16 ? 8:0)
//   storage: pk[((nt*16 + kt)*32 + lane)*16 + j]  (kt contiguous per nt)
// ---------------------------------------------------------------------------
__global__ __launch_bounds__(256) void prep_weights(
    const float* __restrict__ W1, const float* __restrict__ W2,
    _Float16* __restrict__ pk1, _Float16* __restrict__ pk2)
{
  int flat = blockIdx.x * 256 + threadIdx.x;   // 0 .. 2*512*512-1
  int w    = flat >> 18;
  int r    = flat & 0x3FFFF;
  int fragIdx = r >> 9;           // nt*16 + kt
  int within  = r & 511;
  int lane = within >> 4;
  int j    = within & 15;
  int nt = fragIdx >> 4;
  int n  = nt * 16 + (lane & 15);
  int kt = fragIdx & 15;
  int kk = (j & 7) + ((j >> 3) ? 16 : 0) + ((lane >> 4) ? 8 : 0);
  int k  = kt * 32 + kk;
  const float* W  = w ? W2  : W1;
  _Float16*    pk = w ? pk2 : pk1;
  pk[r] = (_Float16)W[n * DIM + k];
}

// ---------------------------------------------------------------------------
// Kernel 1: fused u = tanh(h W1^T + b1); s = u W2^T; per-chunk softmax stats.
// 512 threads = 16 waves. Wave w owns N-tiles {2w, 2w+1} and all 8 M-tiles.
// ---------------------------------------------------------------------------
__global__ __launch_bounds__(512) void fused_attn(
    const float* __restrict__ h, const float* __restrict__ b1v,
    const _Float16* __restrict__ pk1, const _Float16* __restrict__ pk2,
    float* __restrict__ partM, float* __restrict__ partD, float* __restrict__ partN)
{
  __shared__ alignas(32) _Float16 arena[2 * RPB * DIM];  // 256 KB
  _Float16* aFrag = arena;                 // h fragments  (128 KB), dead after GEMM1
  _Float16* uFrag = arena + RPB * DIM;     // u fragments  (128 KB), dead after GEMM2
  float*    sBuf  = (float*)arena;         // scores [128][512] f32, aliases both

  const int tid  = threadIdx.x;            // 0..511
  const int lane = tid & 31;
  const int wave = tid >> 5;               // 0..15
  const int wg   = blockIdx.x;
  const int bidx  = wg >> 5;               // wg / NCHUNK
  const int chunk = wg & (NCHUNK - 1);
  const float* hb = h + ((size_t)bidx * TLEN + (size_t)chunk * RPB) * DIM;

  // ---- stage h tile as f16 A-fragments (32 float4 per thread)
#pragma unroll 4
  for (int i = 0; i < 32; ++i) {
    int f   = tid + i * 512;               // float4 index 0..16383
    int row = f >> 7;                      // 0..127
    int col = (f & 127) << 2;
    float4 v = ((const float4*)hb)[f];
    int mt = row >> 4, m = row & 15;
    int kt = col >> 5, kin = col & 31, sub = kin >> 3;
    int laneA = m + ((sub & 1) << 4);
    int j0 = (kin & 7) + ((sub >> 1) << 3);
    v4h hv = { (_Float16)v.x, (_Float16)v.y, (_Float16)v.z, (_Float16)v.w };
    *(v4h*)&aFrag[FRAG_IDX(kt, mt, laneA) + j0] = hv;
  }
  __syncthreads();

  const int nt0 = wave * 2, nt1 = nt0 + 1;
  const float bias0 = b1v[nt0 * 16 + (lane & 15)];
  const float bias1 = b1v[nt1 * 16 + (lane & 15)];

  // ---- GEMM1: u = tanh(h @ W1^T + b1)   (A loads software-pipelined 1 deep)
  v8f acc[MT][2] = {};
  {
    v16h a = *(const v16h*)&aFrag[FRAG_IDX(0, 0, lane)];
    for (int kt = 0; kt < KT; ++kt) {
      v16h bb0 = *(const v16h*)&pk1[((nt0 * 16 + kt) * 32 + lane) * 16];
      v16h bb1 = *(const v16h*)&pk1[((nt1 * 16 + kt) * 32 + lane) * 16];
#pragma unroll
      for (int mt = 0; mt < MT; ++mt) {
        v16h aCur = a;
        if (!(kt == KT - 1 && mt == MT - 1)) {
          int nkt = (mt == MT - 1) ? kt + 1 : kt;
          int nmt = (mt == MT - 1) ? 0 : mt + 1;
          a = *(const v16h*)&aFrag[FRAG_IDX(nkt, nmt, lane)];
        }
        acc[mt][0] = __builtin_amdgcn_wmma_f32_16x16x32_f16(
            false, aCur, false, bb0, (short)0, acc[mt][0], false, false);
        acc[mt][1] = __builtin_amdgcn_wmma_f32_16x16x32_f16(
            false, aCur, false, bb1, (short)0, acc[mt][1], false, false);
      }
    }
  }

  // epilogue: bias + tanh, scatter into A-fragment layout for GEMM2
#pragma unroll
  for (int q = 0; q < 2; ++q) {
    int e   = (wave * 2 + q) * 16 + (lane & 15);   // GEMM2 K index
    int kt2 = e >> 5, kin = e & 31;
    int laneAdd = ((kin >> 3) & 1) << 4;
    int j = (kin & 7) + (((kin >> 4) & 1) << 3);
    float bias = q ? bias1 : bias0;
#pragma unroll
    for (int mt = 0; mt < MT; ++mt) {
#pragma unroll
      for (int r = 0; r < 8; ++r) {
        int mrow = (lane < 16) ? r : (r + 8);
        float x = acc[mt][q][r] + bias;
        float t = 1.f - 2.f / (__expf(2.f * x) + 1.f);   // tanh
        uFrag[FRAG_IDX(kt2, mt, mrow + laneAdd) + j] = (_Float16)t;
      }
    }
  }
  __syncthreads();

  // ---- GEMM2: s = u @ W2^T
#pragma unroll
  for (int mt = 0; mt < MT; ++mt) { acc[mt][0] = (v8f){}; acc[mt][1] = (v8f){}; }
  {
    v16h a = *(const v16h*)&uFrag[FRAG_IDX(0, 0, lane)];
    for (int kt = 0; kt < KT; ++kt) {
      v16h bb0 = *(const v16h*)&pk2[((nt0 * 16 + kt) * 32 + lane) * 16];
      v16h bb1 = *(const v16h*)&pk2[((nt1 * 16 + kt) * 32 + lane) * 16];
#pragma unroll
      for (int mt = 0; mt < MT; ++mt) {
        v16h aCur = a;
        if (!(kt == KT - 1 && mt == MT - 1)) {
          int nkt = (mt == MT - 1) ? kt + 1 : kt;
          int nmt = (mt == MT - 1) ? 0 : mt + 1;
          a = *(const v16h*)&uFrag[FRAG_IDX(nkt, nmt, lane)];
        }
        acc[mt][0] = __builtin_amdgcn_wmma_f32_16x16x32_f16(
            false, aCur, false, bb0, (short)0, acc[mt][0], false, false);
        acc[mt][1] = __builtin_amdgcn_wmma_f32_16x16x32_f16(
            false, aCur, false, bb1, (short)0, acc[mt][1], false, false);
      }
    }
  }
  __syncthreads();   // all uFrag reads done before sBuf (aliased) is written

#pragma unroll
  for (int q = 0; q < 2; ++q) {
    int e = (wave * 2 + q) * 16 + (lane & 15);
#pragma unroll
    for (int mt = 0; mt < MT; ++mt) {
#pragma unroll
      for (int r = 0; r < 8; ++r) {
        int mrow = (lane < 16) ? r : (r + 8);
        sBuf[(mt * 16 + mrow) * DIM + e] = acc[mt][q][r];
      }
    }
  }
  __syncthreads();

  // ---- per-chunk softmax stats for column `tid` (h re-read from L2, fp32)
  {
    float mx = -1e30f;
    for (int r = 0; r < RPB; ++r)
      mx = fmaxf(mx, sBuf[r * DIM + tid]);
    float den = 0.f, num = 0.f;
    for (int r = 0; r < RPB; ++r) {
      float e = __expf(sBuf[r * DIM + tid] - mx);
      den += e;
      num += hb[(size_t)r * DIM + tid] * e;
    }
    partM[(size_t)wg * DIM + tid] = mx;
    partD[(size_t)wg * DIM + tid] = den;
    partN[(size_t)wg * DIM + tid] = num;
  }
}

// ---------------------------------------------------------------------------
// Kernel 2: merge per-chunk partials per batch (max-rescaled), sum num/den.
// ---------------------------------------------------------------------------
__global__ __launch_bounds__(256) void reduce_out(
    const float* __restrict__ partM, const float* __restrict__ partD,
    const float* __restrict__ partN, float* __restrict__ out)
{
  int d = blockIdx.x * 256 + threadIdx.x;   // 0..511
  float acc = 0.f;
  for (int b = 0; b < BATCH; ++b) {
    float M = -1e30f;
    for (int ch = 0; ch < NCHUNK; ++ch)
      M = fmaxf(M, partM[(size_t)(b * NCHUNK + ch) * DIM + d]);
    float den = 0.f, num = 0.f;
    for (int ch = 0; ch < NCHUNK; ++ch) {
      size_t idx = (size_t)(b * NCHUNK + ch) * DIM + d;
      float sc = __expf(partM[idx] - M);
      den += partD[idx] * sc;
      num += partN[idx] * sc;
    }
    acc += num / den;
  }
  out[d] = acc;
}

extern "C" void kernel_launch(void* const* d_in, const int* in_sizes, int n_in,
                              void* d_out, int out_size, void* d_ws, size_t ws_size,
                              hipStream_t stream) {
  const float* h  = (const float*)d_in[0];
  const float* W1 = (const float*)d_in[1];
  const float* b1 = (const float*)d_in[2];
  const float* W2 = (const float*)d_in[3];
  float* out = (float*)d_out;

  // workspace: 3 partial arrays (2 MB each) + 2 packed f16 weights (512 KB each)
  float* partM = (float*)d_ws;
  float* partD = partM + (size_t)NWG * DIM;
  float* partN = partD + (size_t)NWG * DIM;
  _Float16* pk1 = (_Float16*)(partN + (size_t)NWG * DIM);
  _Float16* pk2 = pk1 + DIM * DIM;

  prep_weights<<<2 * DIM * DIM / 256, 256, 0, stream>>>(W1, W2, pk1, pk2);
  fused_attn<<<NWG, 512, 0, stream>>>(h, b1, pk1, pk2, partM, partD, partN);
  reduce_out<<<DIM / 256, 256, 0, stream>>>(partM, partD, partN, out);
}
